// RBFDD_12025908428900
// MI455X (gfx1250) — matrix operand order repaired and checked by reading, not compiled, not run on hardware
//
#include <hip/hip_runtime.h>
#include <hip/hip_bf16.h>
#include <math.h>

// Problem constants (from reference)
#define N_TOT 16384
#define D_TOT 2048
#define H_TOT 1024

typedef __attribute__((ext_vector_type(16))) __bf16 bf16x16;
typedef __attribute__((ext_vector_type(8)))  __bf16 bf16x8;
typedef __attribute__((ext_vector_type(8)))  float  f32x8;

union FragAB { bf16x16 v; bf16x8 h[2]; };

// ---------------------------------------------------------------------------
// Kernel 1: convert x (fp32, row-major N x D) -> bf16, and x2[n] = sum x^2
// One 256-thread block per row.
// ---------------------------------------------------------------------------
__global__ __launch_bounds__(256)
void prep_x_kernel(const float* __restrict__ x,
                   __bf16* __restrict__ xbf,
                   float* __restrict__ x2)
{
    const int row = blockIdx.x;
    const float* xr  = x   + (size_t)row * D_TOT;
    __bf16*      xbr = xbf + (size_t)row * D_TOT;

    float s = 0.f;
    for (int c = threadIdx.x; c < D_TOT; c += 256) {
        float v = xr[c];
        xbr[c] = (__bf16)v;
        s += v * v;
    }
    // wave32 shuffle reduce, then LDS across the 8 waves
    for (int off = 16; off > 0; off >>= 1)
        s += __shfl_down(s, off, 32);
    __shared__ float red[8];
    if ((threadIdx.x & 31) == 0) red[threadIdx.x >> 5] = s;
    __syncthreads();
    if (threadIdx.x == 0) {
        float t = 0.f;
#pragma unroll
        for (int i = 0; i < 8; i++) t += red[i];
        x2[row] = t;
    }
}

// ---------------------------------------------------------------------------
// Kernel 2: convert Mu -> bf16, mu2[h] = sum Mu^2, coef[h] = 0.5 / Sd^2
// One 256-thread block per row (H rows).
// ---------------------------------------------------------------------------
__global__ __launch_bounds__(256)
void prep_mu_kernel(const float* __restrict__ Mu,
                    const float* __restrict__ Sd,
                    __bf16* __restrict__ mubf,
                    float* __restrict__ mu2,
                    float* __restrict__ coef)
{
    const int row = blockIdx.x;
    const float* mr  = Mu   + (size_t)row * D_TOT;
    __bf16*      mbr = mubf + (size_t)row * D_TOT;

    float s = 0.f;
    for (int c = threadIdx.x; c < D_TOT; c += 256) {
        float v = mr[c];
        mbr[c] = (__bf16)v;
        s += v * v;
    }
    for (int off = 16; off > 0; off >>= 1)
        s += __shfl_down(s, off, 32);
    __shared__ float red[8];
    if ((threadIdx.x & 31) == 0) red[threadIdx.x >> 5] = s;
    __syncthreads();
    if (threadIdx.x == 0) {
        float t = 0.f;
#pragma unroll
        for (int i = 0; i < 8; i++) t += red[i];
        mu2[row] = t;
        float sd = Sd[row];
        coef[row] = 0.5f / (sd * sd);
    }
}

// ---------------------------------------------------------------------------
// Kernel 3: WMMA GEMM + fused RBF epilogue.
//   G = x @ Mu^T   (bf16 inputs, fp32 accumulate, V_WMMA_F32_16X16X32_BF16)
//   p = exp(clip(-(x2 - 2G + mu2) * coef, -100, 40))
//
// Block = 256 threads = 8 wave32s, arranged 2 (M) x 4 (H).
// Each wave computes a 64x64 tile: 4 row-tiles x 4 col-tiles of 16x16
// => 16 WMMAs per K-step against 16 b128 fragment loads (1:1 ratio),
//    each fragment reused by 4 WMMAs.
// Block tile: 128 rows x 256 cols.  Grid: (N/128, H/256) = (128, 4).
//
// Fragment layouts per CDNA5 ISA (CDNA5 ISA):
//   A (16x32 bf16): lane L -> row = L&15; lanes 0-15 hold K 0..7 / 16..23,
//                   lanes 16-31 hold K 8..15 / 24..31 (two 8-elem runs).
//   B (32x16 bf16): lane L -> col = L&15; same K split as A (lanes 0-15
//                   K 0..7 / 16..23, lanes 16-31 K 8..15 / 24..31).
//   C/D (16x16 f32): VGPR r, lane L -> M = r + 8*(L>=16), N = L&15.
// ---------------------------------------------------------------------------
__global__ __launch_bounds__(256)
void rbf_gemm_p_kernel(const __bf16* __restrict__ xbf,
                       const __bf16* __restrict__ mubf,
                       const float* __restrict__ x2,
                       const float* __restrict__ mu2,
                       const float* __restrict__ coef,
                       float* __restrict__ pout)
{
    const int lane   = threadIdx.x & 31;
    const int wave   = threadIdx.x >> 5;
    const int wm     = wave & 1;     // wave position along M (2)
    const int wn     = wave >> 1;    // wave position along H (4)
    const int laneLo = lane & 15;
    const int laneHi = lane >> 4;

    const int m0 = blockIdx.x * 128 + wm * 64;  // first row of this wave
    const int h0 = blockIdx.y * 256 + wn * 64;  // first col of this wave

    f32x8 acc[4][4] = {};

    // Per-lane fragment base pointers (d0 added in the loop)
    const __bf16* aBase[4];
#pragma unroll
    for (int i = 0; i < 4; i++)
        aBase[i] = xbf + (size_t)(m0 + i * 16 + laneLo) * D_TOT + (laneHi << 3);
    const __bf16* bBase[4];
#pragma unroll
    for (int j = 0; j < 4; j++)
        bBase[j] = mubf + (size_t)(h0 + j * 16 + laneLo) * D_TOT + (laneHi << 3);

    for (int d0 = 0; d0 < D_TOT; d0 += 32) {
        FragAB a[4], b[4];
#pragma unroll
        for (int i = 0; i < 4; i++) {
            a[i].h[0] = *(const bf16x8*)(aBase[i] + d0);        // K run 0
            a[i].h[1] = *(const bf16x8*)(aBase[i] + d0 + 16);   // K run 1 (+16)
        }
#pragma unroll
        for (int j = 0; j < 4; j++) {
            b[j].h[0] = *(const bf16x8*)(bBase[j] + d0);        // K run 0
            b[j].h[1] = *(const bf16x8*)(bBase[j] + d0 + 16);   // K run 1 (+16)
        }
#pragma unroll
        for (int i = 0; i < 4; i++)
#pragma unroll
            for (int j = 0; j < 4; j++)
                acc[i][j] = __builtin_amdgcn_wmma_f32_16x16x32_bf16(
                    false, a[i].v, false, b[j].v,
                    (short)0, acc[i][j], false, false);
    }

    // Fused epilogue: dist -> power -> clip -> exp, non-temporal store of p
#pragma unroll
    for (int i = 0; i < 4; i++) {
#pragma unroll
        for (int j = 0; j < 4; j++) {
            const int col = h0 + j * 16 + laneLo;
            const float m2 = mu2[col];
            const float cf = coef[col];
#pragma unroll
            for (int r = 0; r < 8; r++) {
                const int row = m0 + i * 16 + laneHi * 8 + r;
                const float g = acc[i][j][r];
                const float dist = x2[row] - 2.0f * g + m2;
                float pw = -dist * cf;
                pw = fminf(fmaxf(pw, -100.0f), 40.0f);
                const float pv = expf(pw);
                __builtin_nontemporal_store(pv, &pout[(size_t)row * H_TOT + col]);
            }
        }
    }
}

// ---------------------------------------------------------------------------
// Kernel 4: z = p @ W, y = 1.7159 * tanh((2/3) z).  One block per row.
// ---------------------------------------------------------------------------
__global__ __launch_bounds__(256)
void zy_kernel(const float* __restrict__ p,
               const float* __restrict__ W,
               float* __restrict__ y)
{
    const int row = blockIdx.x;
    const float* pr = p + (size_t)row * H_TOT;
    float s = 0.f;
    for (int c = threadIdx.x; c < H_TOT; c += 256)
        s += pr[c] * W[c];
    for (int off = 16; off > 0; off >>= 1)
        s += __shfl_down(s, off, 32);
    __shared__ float red[8];
    if ((threadIdx.x & 31) == 0) red[threadIdx.x >> 5] = s;
    __syncthreads();
    if (threadIdx.x == 0) {
        float t = 0.f;
#pragma unroll
        for (int i = 0; i < 8; i++) t += red[i];
        y[row] = 1.7159f * tanhf((2.0f / 3.0f) * t);
    }
}

// ---------------------------------------------------------------------------
// Host launcher
// ---------------------------------------------------------------------------
extern "C" void kernel_launch(void* const* d_in, const int* in_sizes, int n_in,
                              void* d_out, int out_size, void* d_ws, size_t ws_size,
                              hipStream_t stream)
{
    const float* x  = (const float*)d_in[0];   // (N, D)
    const float* Mu = (const float*)d_in[1];   // (H, D)
    const float* Sd = (const float*)d_in[2];   // (H,)
    const float* W  = (const float*)d_in[3];   // (1, H)

    float* y = (float*)d_out;                  // (N,)  first output
    float* p = (float*)d_out + N_TOT;          // (N,H) second output

    // Workspace carve-up (all offsets 16B-aligned)
    char* ws = (char*)d_ws;
    __bf16* xbf  = (__bf16*)ws;                        ws += (size_t)N_TOT * D_TOT * 2;
    __bf16* mubf = (__bf16*)ws;                        ws += (size_t)H_TOT * D_TOT * 2;
    float*  x2   = (float*)ws;                         ws += (size_t)N_TOT * 4;
    float*  mu2  = (float*)ws;                         ws += (size_t)H_TOT * 4;
    float*  coef = (float*)ws;                         ws += (size_t)H_TOT * 4;

    prep_x_kernel<<<N_TOT, 256, 0, stream>>>(x, xbf, x2);
    prep_mu_kernel<<<H_TOT, 256, 0, stream>>>(Mu, Sd, mubf, mu2, coef);

    dim3 grid(N_TOT / 128, H_TOT / 256);  // (128, 4)
    rbf_gemm_p_kernel<<<grid, 256, 0, stream>>>(xbf, mubf, x2, mu2, coef, p);

    zy_kernel<<<N_TOT, 256, 0, stream>>>(p, W, y);
}
